// HCGN_64768106823900
// MI455X (gfx1250) — matrix operand order repaired and checked by reading, not compiled, not run on hardware
//
#include <hip/hip_runtime.h>
#include <hip/hip_bf16.h>

// ---------------------------------------------------------------------------
// Problem constants (match reference)
// ---------------------------------------------------------------------------
#define NN   8192      // nodes
#define CC   128       // input dim
#define TT   256       // theta_1 dim
#define OO   128       // output dim
#define NEGV (-9e15f)
#define SLOPE 0.01f

typedef __attribute__((ext_vector_type(16))) __bf16 v16bf;
typedef __attribute__((ext_vector_type(8)))  float  v8f;
typedef __attribute__((ext_vector_type(4)))  float  vf4;

union FragB16 { v16bf v; uint4 q[2]; };

// A-matrix 16x32 bf16 fragment (ISA 7.12.2): lane L holds row M=L%16;
// lanes 0-15: K = {k0+0..7, k0+16..23}; lanes 16-31: K = {k0+8..15, k0+24..31}.
__device__ __forceinline__ v16bf load_a_frag_bf16(const __bf16* base, int row0,
                                                  int ld, int k0, int lane) {
    int m  = lane & 15;
    int kh = k0 + ((lane >> 4) << 3);
    const __bf16* p = base + (size_t)(row0 + m) * ld + kh;
    FragB16 f;
    f.q[0] = *(const uint4*)(p);        // K = kh .. kh+7
    f.q[1] = *(const uint4*)(p + 16);   // K = kh+16 .. kh+23
    return f.v;
}

// B-matrix 32x16 bf16 fragment: lane L holds column N=L%16;
// lanes 0-15: K = {k0+0..7, k0+16..23}; lanes 16-31: K = {k0+8..15, k0+24..31}.
// `base` is column-major over K (i.e. row n of `base` holds B[:,n]).
__device__ __forceinline__ v16bf load_b_frag_bf16(const __bf16* base, int col0,
                                                  int ld, int k0, int lane) {
    int n  = lane & 15;
    int ks = k0 + ((lane >> 4) << 3);
    const __bf16* p = base + (size_t)(col0 + n) * ld + ks;
    FragB16 f;
    f.q[0] = *(const uint4*)(p);
    f.q[1] = *(const uint4*)(p + 16);
    return f.v;
}

// ---------------------------------------------------------------------------
// 1) BatchNorm statistics
// ---------------------------------------------------------------------------
__global__ void zero_stats_kernel(float* stats) {          // 1 block, 256 thr
    stats[threadIdx.x] = 0.0f;                              // colsum+colsq
}

__global__ __launch_bounds__(256) void bn_partial_kernel(const float* __restrict__ H,
                                                         float* colsum, float* colsq) {
    int t  = threadIdx.x;
    int c  = t & (CC - 1);
    int rh = t >> 7;                       // 0 or 1
    int r0 = blockIdx.x * 64;
    float s = 0.f, ss = 0.f;
    for (int r = rh; r < 64; r += 2) {
        float x = H[(size_t)(r0 + r) * CC + c];
        s += x; ss += x * x;
    }
    atomicAdd(&colsum[c], s);
    atomicAdd(&colsq[c], ss);
}

__global__ void bn_finalize_kernel(float* colsum, float* colsq) {  // 1 blk, 128 thr
    int c = threadIdx.x;
    float mu  = colsum[c] / (float)NN;
    float var = colsq[c] / (float)NN - mu * mu;
    colsum[c] = mu;
    colsq[c]  = rsqrtf(var + 1e-5f);
}

// ---------------------------------------------------------------------------
// 2) BN-normalize + theta_1 / W_out projections (bf16 outputs, HW transposed)
// ---------------------------------------------------------------------------
__global__ __launch_bounds__(256) void project_kernel(
        const float* __restrict__ H,  const float* __restrict__ mu,
        const float* __restrict__ rsig, const float* __restrict__ gamma,
        const float* __restrict__ beta, const float* __restrict__ W1,
        const float* __restrict__ b1,   const float* __restrict__ Wo,
        const float* __restrict__ bo,   __bf16* __restrict__ Hx,
        __bf16* __restrict__ HWt) {
    __shared__ float hn[CC];
    int i = blockIdx.x;
    int t = threadIdx.x;
    if (t < CC) {
        float x = H[(size_t)i * CC + t];
        hn[t] = (x - mu[t]) * rsig[t] * gamma[t] + beta[t];
    }
    __syncthreads();
    {   // Hx[i, t], t in [0,256)
        float acc = b1[t];
        #pragma unroll 4
        for (int c = 0; c < CC; ++c) acc = fmaf(hn[c], W1[c * TT + t], acc);
        Hx[(size_t)i * TT + t] = (__bf16)acc;
    }
    if (t < OO) {   // HWt[t, i] = (Hn @ Wo + bo)[i, t]  (transposed, bf16)
        float acc = bo[t];
        #pragma unroll 4
        for (int c = 0; c < CC; ++c) acc = fmaf(hn[c], Wo[c * OO + t], acc);
        HWt[(size_t)t * NN + i] = (__bf16)acc;
    }
}

// ---------------------------------------------------------------------------
// 3) scores: A1 = where(A>0, sigmoid(Hx @ Hx^T), NEG)   [WMMA bf16]
//    Block: 128x128 tile, 8 waves as 4x2 grid; wave = 32x64 (2x4 WMMA tiles).
//    Mask A is single-use -> non-temporal loads; A1 is re-read twice -> cached.
// ---------------------------------------------------------------------------
__global__ __launch_bounds__(256) void scores_kernel(const __bf16* __restrict__ Hx,
                                                     const float* __restrict__ A,
                                                     float* __restrict__ A1) {
    const int lane = threadIdx.x & 31;
    const int w    = threadIdx.x >> 5;
    const int wr   = w >> 1, wc = w & 1;
    const int Rb   = blockIdx.y * 128 + wr * 32;
    const int Cb   = blockIdx.x * 128 + wc * 64;

    v8f acc[2][4];
    const v8f vz = {0.f,0.f,0.f,0.f,0.f,0.f,0.f,0.f};
    #pragma unroll
    for (int i = 0; i < 2; ++i)
        #pragma unroll
        for (int j = 0; j < 4; ++j) acc[i][j] = vz;

    for (int k0 = 0; k0 < TT; k0 += 32) {
        v16bf af[2], bf[4];
        #pragma unroll
        for (int i = 0; i < 2; ++i)
            af[i] = load_a_frag_bf16(Hx, Rb + i * 16, TT, k0, lane);
        #pragma unroll
        for (int j = 0; j < 4; ++j)
            bf[j] = load_b_frag_bf16(Hx, Cb + j * 16, TT, k0, lane);
        #pragma unroll
        for (int i = 0; i < 2; ++i)
            #pragma unroll
            for (int j = 0; j < 4; ++j)
                acc[i][j] = __builtin_amdgcn_wmma_f32_16x16x32_bf16(
                    false, af[i], false, bf[j], (short)0, acc[i][j], false, false);
    }

    // Epilogue: fast sigmoid (v_rcp_f32) + mask, write A1 (C-layout, ISA 7.12.2).
    const int rhalf = (lane >> 4) << 3;      // 0 or 8
    const int col0  = lane & 15;
    #pragma unroll
    for (int i = 0; i < 2; ++i) {
        #pragma unroll
        for (int j = 0; j < 4; ++j) {
            int c = Cb + j * 16 + col0;
            #pragma unroll
            for (int e = 0; e < 8; ++e) {
                int row = Rb + i * 16 + rhalf + e;
                float s  = acc[i][j][e];
                float ev = __builtin_amdgcn_rcpf(1.f + __expf(-s));
                float mk = __builtin_nontemporal_load(&A[(size_t)row * NN + c]);
                A1[(size_t)row * NN + c] = (mk > 0.f) ? ev : NEGV;
            }
        }
    }
}

// ---------------------------------------------------------------------------
// 4) rowstats: online (flash-style) row max/sum for both softmax variants.
//    One wave per row; variants differ only at the diagonal (+1 from eye).
// ---------------------------------------------------------------------------
__global__ __launch_bounds__(256) void rowstats_kernel(const float* __restrict__ A1,
        float* __restrict__ m1a, float* __restrict__ il1a,
        float* __restrict__ m2a, float* __restrict__ il2a) {
    const int lane = threadIdx.x & 31;
    const int row  = blockIdx.x * 8 + (threadIdx.x >> 5);
    const float* pr = A1 + (size_t)row * NN;

    float lm = -3.0e38f, ls = 0.f, vd = -3.0e38f;
    for (int j = lane; j < NN; j += 32) {
        float a = pr[j];
        if (j == row) vd = a;
        if (a > lm) { ls = ls * __expf(lm - a) + 1.f; lm = a; }
        else        { ls += __expf(a - lm); }
    }
    #pragma unroll
    for (int off = 16; off >= 1; off >>= 1) {
        float om = __shfl_xor(lm, off, 32);
        float os = __shfl_xor(ls, off, 32);
        float nm = fmaxf(lm, om);
        ls = ls * __expf(lm - nm) + os * __expf(om - nm);
        lm = nm;
        vd = fmaxf(vd, __shfl_xor(vd, off, 32));
    }
    if (lane == 0) {
        float m2 = lm, l2 = ls;
        float m1 = fmaxf(m2, vd + 1.f);
        float l1 = __expf(m2 - m1) * l2 - __expf(vd - m1) + __expf(vd + 1.f - m1);
        m1a[row] = m1;  il1a[row] = 1.f / l1;
        m2a[row] = m2;  il2a[row] = 1.f / l2;
    }
}

// ---------------------------------------------------------------------------
// 5) attend: finalize A_soft in place and compute out = leaky(P1@HW)+leaky(P2@HW).
//    Wave = 16 rows x 128 out-cols; K (=node index) swept in steps of 32.
//    Each A1 element is read exactly once by the lane that overwrites it (safe).
//    A_soft write-back is final output (never re-read) -> non-temporal stores.
// ---------------------------------------------------------------------------
__global__ __launch_bounds__(256) void attend_kernel(float* __restrict__ A1,
        const __bf16* __restrict__ HWt,
        const float* __restrict__ m1a, const float* __restrict__ il1a,
        const float* __restrict__ m2a, const float* __restrict__ il2a,
        float* __restrict__ out) {
    const int lane = threadIdx.x & 31;
    const int w    = threadIdx.x >> 5;
    const int Rb   = blockIdx.x * 128 + w * 16;
    const int mrow = Rb + (lane & 15);

    const float M1 = m1a[mrow], IL1 = il1a[mrow];
    const float M2 = m2a[mrow], IL2 = il2a[mrow];

    v8f acc1[8], acc2[8];
    const v8f vz = {0.f,0.f,0.f,0.f,0.f,0.f,0.f,0.f};
    #pragma unroll
    for (int ct = 0; ct < 8; ++ct) { acc1[ct] = vz; acc2[ct] = vz; }

    for (int kb = 0; kb < NN; kb += 32) {
        const int kh = kb + ((lane >> 4) << 3);
        float* pw = A1 + (size_t)mrow * NN + kh;
        vf4 f0 = *(const vf4*)(pw);
        vf4 f1 = *(const vf4*)(pw + 4);
        vf4 f2 = *(const vf4*)(pw + 16);
        vf4 f3 = *(const vf4*)(pw + 20);
        float vals[16] = {f0.x,f0.y,f0.z,f0.w, f1.x,f1.y,f1.z,f1.w,
                          f2.x,f2.y,f2.z,f2.w, f3.x,f3.y,f3.z,f3.w};
        FragB16 fr1, fr2;
        float p1s[16];
        #pragma unroll
        for (int e = 0; e < 16; ++e) {
            int   j   = kh + (e < 8 ? e : e + 8);
            float a   = vals[e];
            float a1v = a + ((j == mrow) ? 1.f : 0.f);     // + eye
            float p1  = __expf(a1v - M1) * IL1;
            float p2  = __expf(a   - M2) * IL2;
            fr1.v[e] = (__bf16)p1;
            fr2.v[e] = (__bf16)p2;
            p1s[e] = p1;
        }
        // In-place finalize A_soft (each address: this lane read it above).
        vf4 o0 = {p1s[0],  p1s[1],  p1s[2],  p1s[3]};
        vf4 o1 = {p1s[4],  p1s[5],  p1s[6],  p1s[7]};
        vf4 o2 = {p1s[8],  p1s[9],  p1s[10], p1s[11]};
        vf4 o3 = {p1s[12], p1s[13], p1s[14], p1s[15]};
        __builtin_nontemporal_store(o0, (vf4*)(pw));
        __builtin_nontemporal_store(o1, (vf4*)(pw + 4));
        __builtin_nontemporal_store(o2, (vf4*)(pw + 16));
        __builtin_nontemporal_store(o3, (vf4*)(pw + 20));

        #pragma unroll
        for (int ct = 0; ct < 8; ++ct) {
            v16bf b = load_b_frag_bf16(HWt, ct * 16, NN, kb, lane);
            acc1[ct] = __builtin_amdgcn_wmma_f32_16x16x32_bf16(
                false, fr1.v, false, b, (short)0, acc1[ct], false, false);
            acc2[ct] = __builtin_amdgcn_wmma_f32_16x16x32_bf16(
                false, fr2.v, false, b, (short)0, acc2[ct], false, false);
        }
    }

    const int rhalf = (lane >> 4) << 3;
    const int ocol  = lane & 15;
    #pragma unroll
    for (int ct = 0; ct < 8; ++ct) {
        #pragma unroll
        for (int e = 0; e < 8; ++e) {
            int row = Rb + rhalf + e;
            int o   = ct * 16 + ocol;
            float x1 = acc1[ct][e];
            float x2 = acc2[ct][e];
            float y1 = (x1 >= 0.f) ? x1 : SLOPE * x1;
            float y2 = (x2 >= 0.f) ? x2 : SLOPE * x2;
            out[(size_t)row * OO + o] = y1 + y2;
        }
    }
}

// ---------------------------------------------------------------------------
// Launcher
// ---------------------------------------------------------------------------
extern "C" void kernel_launch(void* const* d_in, const int* in_sizes, int n_in,
                              void* d_out, int out_size, void* d_ws, size_t ws_size,
                              hipStream_t stream) {
    const float* H     = (const float*)d_in[0];
    const float* A     = (const float*)d_in[1];
    const float* gamma = (const float*)d_in[2];
    const float* beta  = (const float*)d_in[3];
    const float* W1    = (const float*)d_in[4];
    const float* b1    = (const float*)d_in[5];
    const float* Wo    = (const float*)d_in[6];
    const float* bo    = (const float*)d_in[7];

    float* out_ptr = (float*)d_out;                     // [N, OUT]
    float* A_soft  = (float*)d_out + (size_t)NN * OO;   // [N, N]; also A1 scratch

    // Workspace layout (floats): colsum[128] colsq[128] m1[N] il1[N] m2[N] il2[N]
    // then bf16 Hx [N,T] and bf16 HWt [OUT,N].  Total ~6.4 MB.
    float*  wsf    = (float*)d_ws;
    float*  colsum = wsf;
    float*  colsq  = wsf + CC;
    float*  m1a    = wsf + 2 * CC;
    float*  il1a   = m1a + NN;
    float*  m2a    = il1a + NN;
    float*  il2a   = m2a + NN;
    __bf16* Hx     = (__bf16*)(il2a + NN);
    __bf16* HWt    = Hx + (size_t)NN * TT;

    zero_stats_kernel<<<1, 2 * CC, 0, stream>>>(colsum);
    bn_partial_kernel<<<NN / 64, 256, 0, stream>>>(H, colsum, colsq);
    bn_finalize_kernel<<<1, CC, 0, stream>>>(colsum, colsq);
    project_kernel<<<NN, 256, 0, stream>>>(H, colsum, colsq, gamma, beta,
                                           W1, b1, Wo, bo, Hx, HWt);
    scores_kernel<<<dim3(NN / 128, NN / 128), 256, 0, stream>>>(Hx, A, A_soft);
    rowstats_kernel<<<NN / 8, 256, 0, stream>>>(A_soft, m1a, il1a, m2a, il2a);
    attend_kernel<<<NN / 128, 256, 0, stream>>>(A_soft, HWt, m1a, il1a, m2a, il2a,
                                                out_ptr);
}
